// GINLayer_19172734010018
// MI455X (gfx1250) — matrix operand order repaired and checked by reading, not compiled, not run on hardware
//
#include <hip/hip_runtime.h>
#include <hip/hip_bf16.h>

typedef __attribute__((ext_vector_type(16))) __bf16 v16bf;
typedef __attribute__((ext_vector_type(8)))  __bf16 v8bf;
typedef __attribute__((ext_vector_type(8)))  float  v8f;

constexpr int N_NODES = 50000;
constexpr int D       = 512;        // D_IN == D_OUT == 512
constexpr int N_EDGES = 800000;
constexpr int MTILES  = N_NODES / 16;   // 3125 (exact)
constexpr int NTILES  = D / 16;         // 32

// ---------------------------------------------------------------- h = x (init accumulator)
__global__ __launch_bounds__(256) void gin_copy_x(const float4* __restrict__ x,
                                                  float4* __restrict__ h) {
    int i = blockIdx.x * 256 + threadIdx.x;
    if (i < N_NODES * D / 4) h[i] = x[i];
}

// ------------------------------------------- Wt[n][k] = bf16(W[k][n]) for W1 and W2
__global__ __launch_bounds__(256) void gin_convert_w(const float* __restrict__ W1,
                                                     const float* __restrict__ W2,
                                                     __bf16* __restrict__ W1t,
                                                     __bf16* __restrict__ W2t) {
    int i = blockIdx.x * 256 + threadIdx.x;        // 0 .. 2*D*D-1
    int which = i >= D * D;
    int j = i - which * D * D;
    int n = j / D, k = j % D;
    const float* W = which ? W2 : W1;
    __bf16*     Wt = which ? W2t : W1t;
    Wt[(size_t)n * D + k] = (__bf16)W[(size_t)k * D + n];
}

// ---------------------------------------------------------------- scatter-add: h[dst] += x[src]
__global__ __launch_bounds__(256) void gin_scatter(const float* __restrict__ x,
                                                   const long long* __restrict__ ei,
                                                   float* __restrict__ h) {
    int wave = threadIdx.x >> 5;
    int lane = threadIdx.x & 31;
    int e = blockIdx.x * 8 + wave;
    if (e >= N_EDGES) return;
    int src = (int)ei[e];
    int dst = (int)ei[N_EDGES + e];
    const float4* xs = (const float4*)(x + (size_t)src * D);
    float*        hd = h + (size_t)dst * D;
#pragma unroll
    for (int i = 0; i < 4; ++i) {
        int c4 = i * 32 + lane;                    // 128 float4 per row, coalesced
        float4 v = xs[c4];
        unsafeAtomicAdd(hd + c4 * 4 + 0, v.x);
        unsafeAtomicAdd(hd + c4 * 4 + 1, v.y);
        unsafeAtomicAdd(hd + c4 * 4 + 2, v.z);
        unsafeAtomicAdd(hd + c4 * 4 + 3, v.w);
    }
}

// ---------------------------------------------------------------- WMMA GEMM + bias + ReLU
// One wave -> one 16x16 C tile. A: [M,512] (fp32 or bf16 row-major), B: Wt[n][k] bf16.
// ISA 16-bit A layout: lane<16 -> row=lane, K groups {0..7, 16..23}; lane>=16 -> row=lane-16,
// K groups {8..15, 24..31}. B (K-major per column): lane<16 -> col=lane, K {0..7, 16..23};
// lane>=16 -> col=lane-16, K {8..15, 24..31}. C: VGPR r -> row r + 8*(lane>=16), col = lane&15.
template <bool A_BF16, bool OUT_FP32>
__global__ __launch_bounds__(256) void gin_gemm(const void* __restrict__ Ain,
                                                const __bf16* __restrict__ Wt,
                                                const float* __restrict__ bias,
                                                void* __restrict__ Yout) {
    int wave = threadIdx.x >> 5;
    int lane = threadIdx.x & 31;
    int mtile = blockIdx.x >> 2;                     // 4 blocks per M tile
    int ntile = ((blockIdx.x & 3) << 3) + wave;      // 8 waves cover 8 N tiles
    int m0 = mtile * 16;
    int n0 = ntile * 16;

    int row  = lane & 15;
    int half = lane >> 4;                            // 0 or 1
    int col  = n0 + row;
    int akb  = half * 8;                             // A K-group base
    int bkb  = half * 8;                             // B K-group base

    v8f c;
    float bv = bias[col];
#pragma unroll
    for (int r = 0; r < 8; ++r) c[r] = bv;

    const __bf16* bptr = Wt + (size_t)col * D + bkb;

    for (int kk = 0; kk < D; kk += 32) {
        v16bf a;
        if (A_BF16) {
            const __bf16* arow = (const __bf16*)Ain + (size_t)(m0 + row) * D + kk + akb;
            v8bf g0 = *(const v8bf*)(arow);          // K akb .. akb+7
            v8bf g1 = *(const v8bf*)(arow + 16);     // K akb+16 .. akb+23
#pragma unroll
            for (int r = 0; r < 8; ++r) { a[r] = g0[r]; a[8 + r] = g1[r]; }
        } else {
            const float* arow = (const float*)Ain + (size_t)(m0 + row) * D + kk + akb;
            float4 a0 = *(const float4*)(arow);
            float4 a1 = *(const float4*)(arow + 4);
            float4 a2 = *(const float4*)(arow + 16);
            float4 a3 = *(const float4*)(arow + 20);
            a[0]=(__bf16)a0.x;  a[1]=(__bf16)a0.y;  a[2]=(__bf16)a0.z;  a[3]=(__bf16)a0.w;
            a[4]=(__bf16)a1.x;  a[5]=(__bf16)a1.y;  a[6]=(__bf16)a1.z;  a[7]=(__bf16)a1.w;
            a[8]=(__bf16)a2.x;  a[9]=(__bf16)a2.y;  a[10]=(__bf16)a2.z; a[11]=(__bf16)a2.w;
            a[12]=(__bf16)a3.x; a[13]=(__bf16)a3.y; a[14]=(__bf16)a3.z; a[15]=(__bf16)a3.w;
        }
        v8bf b0 = *(const v8bf*)(bptr + kk);
        v8bf b1 = *(const v8bf*)(bptr + kk + 16);
        v16bf b;
#pragma unroll
        for (int r = 0; r < 8; ++r) { b[r] = b0[r]; b[8 + r] = b1[r]; }
        c = __builtin_amdgcn_wmma_f32_16x16x32_bf16(false, a, false, b,
                                                    (short)0, c, false, false);
    }

    int mbase = m0 + half * 8;
    if (OUT_FP32) {
        float* out = (float*)Yout;
#pragma unroll
        for (int r = 0; r < 8; ++r)
            out[(size_t)(mbase + r) * D + col] = fmaxf(c[r], 0.0f);
    } else {
        __bf16* out = (__bf16*)Yout;
#pragma unroll
        for (int r = 0; r < 8; ++r)
            out[(size_t)(mbase + r) * D + col] = (__bf16)fmaxf(c[r], 0.0f);
    }
}

extern "C" void kernel_launch(void* const* d_in, const int* in_sizes, int n_in,
                              void* d_out, int out_size, void* d_ws, size_t ws_size,
                              hipStream_t stream) {
    const float*     x  = (const float*)d_in[0];
    const long long* ei = (const long long*)d_in[1];
    const float*     W1 = (const float*)d_in[2];
    const float*     b1 = (const float*)d_in[3];
    const float*     W2 = (const float*)d_in[4];
    const float*     b2 = (const float*)d_in[5];
    float* out = (float*)d_out;

    char* ws = (char*)d_ws;
    __bf16* Y   = (__bf16*)ws;                                   // 50000*512 bf16 = 51.2 MB
    __bf16* W1t = (__bf16*)(ws + (size_t)N_NODES * D * 2);       // 512*512 bf16
    __bf16* W2t = W1t + (size_t)D * D;

    // 1) h := x  (accumulate in d_out; fully rewritten every call -> deterministic)
    gin_copy_x<<<(N_NODES * D / 4 + 255) / 256, 256, 0, stream>>>((const float4*)x,
                                                                  (float4*)out);
    // 2) weights -> bf16 transposed
    gin_convert_w<<<(2 * D * D) / 256, 256, 0, stream>>>(W1, W2, W1t, W2t);
    // 3) h += scatter(x[src] -> dst)      (L2-resident atomics)
    gin_scatter<<<(N_EDGES + 7) / 8, 256, 0, stream>>>(x, ei, out);
    // 4) Y = relu(h @ W1 + b1)  -> bf16
    gin_gemm<false, false><<<MTILES * 4, 256, 0, stream>>>((const void*)out, W1t, b1, (void*)Y);
    // 5) out = relu(Y @ W2 + b2) -> fp32
    gin_gemm<true, true><<<MTILES * 4, 256, 0, stream>>>((const void*)Y, W2t, b2, (void*)out);
}
